// DINAttention_11252814315850
// MI455X (gfx1250) — matrix operand order repaired and checked by reading, not compiled, not run on hardware
//
#include <hip/hip_runtime.h>
#include <hip/hip_bf16.h>
#include <cstdint>

typedef __attribute__((ext_vector_type(16))) _Float16 v16h;
typedef __attribute__((ext_vector_type(8)))  _Float16 v8h;
typedef __attribute__((ext_vector_type(8)))  float    v8f;
typedef __attribute__((ext_vector_type(4)))  float    f4v;

#define B_   2048
#define L_   200
#define D_   128
#define K1_  512     // 4*D concat width
#define H1_  256
#define H2_  128
#define LP_  224     // padded L (7 double-tiles of 32)
#define NT_  7
#define NEGV (-10000.0f)

static __device__ inline v16h cat8(v8h lo, v8h hi) {
  return __builtin_shufflevector(lo, hi, 0,1,2,3,4,5,6,7,8,9,10,11,12,13,14,15);
}

// ---------------------------------------------------------------------------
// Prep: W1[512,256] f32 -> W1T[256][512] f16 (N-major), W2[256,128] -> W2T[128][256]
// ---------------------------------------------------------------------------
__global__ void din_prep_weights(const float* __restrict__ W1,
                                 const float* __restrict__ W2,
                                 _Float16* __restrict__ W1T,
                                 _Float16* __restrict__ W2T) {
  int idx = blockIdx.x * 256 + threadIdx.x;
  if (idx < H1_ * K1_) {                 // 131072
    int n = idx >> 9;                    // /512
    int k = idx & (K1_ - 1);
    W1T[idx] = (_Float16)W1[k * H1_ + n];
  }
  if (idx < H2_ * H1_) {                 // 32768
    int n = idx >> 8;                    // /256
    int k = idx & (H1_ - 1);
    W2T[idx] = (_Float16)W2[k * H2_ + n];
  }
}

// ---------------------------------------------------------------------------
// Main: one workgroup (8 wave32) per batch element; 32 tokens per iteration
// ---------------------------------------------------------------------------
__launch_bounds__(256, 1)
__global__ void din_attention_kernel(
    const float* __restrict__ query, const float* __restrict__ keys,
    const int*   __restrict__ mask,
    const float* __restrict__ b1, const float* __restrict__ a1,
    const float* __restrict__ b2, const float* __restrict__ a2,
    const float* __restrict__ W3, const float* __restrict__ b3,
    const _Float16* __restrict__ W1T, const _Float16* __restrict__ W2T,
    float* __restrict__ out)
{
  // ---- LDS (≈322 KB of the 327,680 B WGP budget) ----
  __shared__ __align__(16) _Float16 sW1T[H1_ * K1_];   // 262144 B
  __shared__ __align__(16) _Float16 sAtt[32 * K1_];    //  32768 B
  __shared__ __align__(16) _Float16 sH1 [32 * H1_];    //  16384 B
  __shared__ __align__(16) _Float16 sH2 [32 * H2_];    //   8192 B
  __shared__ float sQ[D_];
  __shared__ float sAcc[32];
  __shared__ float sScr[LP_];
  __shared__ float sPrb[LP_];
  __shared__ float sRed[9];

  const int tid  = threadIdx.x;
  const int wave = tid >> 5;
  const int lane = tid & 31;
  const int col  = lane & 15;   // N column within a 16-wide tile; also A-row
  const int half = lane >> 4;   // K-half selector per WMMA register layout
  const int b    = blockIdx.x;

  // ---- stage W1T into LDS: 16384 x 16B chunks, 64 per thread ----
  {
    const f4v* src = (const f4v*)W1T;
    f4v*       dst = (f4v*)sW1T;
    #pragma unroll 8
    for (int i = 0; i < 64; ++i)
      dst[i * 256 + tid] = src[i * 256 + tid];
  }
  if (tid < D_) sQ[tid] = query[(size_t)b * D_ + tid];
  __syncthreads();

  const float pa1 = a1[0], pa2 = a2[0], pb3 = b3[0];

  for (int t = 0; t < NT_; ++t) {
    const int l0 = t * 32;

    // ---- build att_input tile [32 x 512] f16: [q | k | q-k | q*k] ----
    #pragma unroll
    for (int it = 0; it < 8; ++it) {
      int idx = tid + it * 256;       // 0..2047 = 32 rows * 64 d-pairs
      int d2  = idx & 63;
      int row = idx >> 6;
      int d   = d2 * 2;
      int l   = l0 + row;
      float k0 = 0.f, k1 = 0.f;
      if (l < L_) {
        const float* kp = keys + (((size_t)b * L_ + l) * D_ + d);
        k0 = kp[0]; k1 = kp[1];
      }
      float q0 = sQ[d], q1 = sQ[d + 1];
      _Float16* ap = sAtt + row * K1_;
      ap[d]       = (_Float16)q0;        ap[d + 1]   = (_Float16)q1;
      ap[128 + d] = (_Float16)k0;        ap[129 + d] = (_Float16)k1;
      ap[256 + d] = (_Float16)(q0 - k0); ap[257 + d] = (_Float16)(q1 - k1);
      ap[384 + d] = (_Float16)(q0 * k0); ap[385 + d] = (_Float16)(q1 * k1);
    }
    __syncthreads();

    // ---- layer 1: [32x512] @ [512x256] -> h1[32x256]
    //      wave owns 2 N-tiles x 2 row-tiles: B reused over rows, A over cols ----
    {
      const int n0 = wave * 32 + col;
      const int n1 = n0 + 16;
      v8f acc00, acc01, acc10, acc11;
      const float bv0 = b1[n0], bv1 = b1[n1];
      #pragma unroll
      for (int r = 0; r < 8; ++r) {
        acc00[r] = bv0; acc01[r] = bv1;   // row-tile 0
        acc10[r] = bv0; acc11[r] = bv1;   // row-tile 1
      }

      const _Float16* arow0 = sAtt + col * K1_;          // rows 0..15
      const _Float16* arow1 = sAtt + (16 + col) * K1_;   // rows 16..31
      #pragma unroll 2
      for (int kk = 0; kk < 16; ++kk) {
        const int kb = kk * 32;
        v16h a0 = cat8(*(const v8h*)(arow0 + kb + half * 8),
                       *(const v8h*)(arow0 + kb + 16 + half * 8));
        v16h a1v = cat8(*(const v8h*)(arow1 + kb + half * 8),
                        *(const v8h*)(arow1 + kb + 16 + half * 8));

        const _Float16* bp0 = sW1T + n0 * K1_ + kb + half * 8;
        v16h bm0 = cat8(*(const v8h*)bp0, *(const v8h*)(bp0 + 16));
        const _Float16* bp1 = sW1T + n1 * K1_ + kb + half * 8;
        v16h bm1 = cat8(*(const v8h*)bp1, *(const v8h*)(bp1 + 16));

        acc00 = __builtin_amdgcn_wmma_f32_16x16x32_f16(false, a0,  false, bm0,
                                                       (short)0, acc00, false, false);
        acc01 = __builtin_amdgcn_wmma_f32_16x16x32_f16(false, a0,  false, bm1,
                                                       (short)0, acc01, false, false);
        acc10 = __builtin_amdgcn_wmma_f32_16x16x32_f16(false, a1v, false, bm0,
                                                       (short)0, acc10, false, false);
        acc11 = __builtin_amdgcn_wmma_f32_16x16x32_f16(false, a1v, false, bm1,
                                                       (short)0, acc11, false, false);
      }
      #pragma unroll
      for (int r = 0; r < 8; ++r) {
        int row0 = r + half * 8;
        int row1 = 16 + row0;
        float x;
        x = acc00[r]; x = (x >= 0.f) ? x : pa1 * x; sH1[row0 * H1_ + n0] = (_Float16)x;
        x = acc01[r]; x = (x >= 0.f) ? x : pa1 * x; sH1[row0 * H1_ + n1] = (_Float16)x;
        x = acc10[r]; x = (x >= 0.f) ? x : pa1 * x; sH1[row1 * H1_ + n0] = (_Float16)x;
        x = acc11[r]; x = (x >= 0.f) ? x : pa1 * x; sH1[row1 * H1_ + n1] = (_Float16)x;
      }
    }
    __syncthreads();

    // ---- layer 2: [32x256] @ [256x128] -> h2[32x128]; wave owns 16 cols x 2 row-tiles ----
    {
      const int n = wave * 16 + col;
      v8f acc0, acc1;
      const float bv = b2[n];
      #pragma unroll
      for (int r = 0; r < 8; ++r) { acc0[r] = bv; acc1[r] = bv; }

      const _Float16* arow0 = sH1 + col * H1_;
      const _Float16* arow1 = sH1 + (16 + col) * H1_;
      #pragma unroll 4
      for (int kk = 0; kk < 8; ++kk) {
        const int kb = kk * 32;
        v16h a0 = cat8(*(const v8h*)(arow0 + kb + half * 8),
                       *(const v8h*)(arow0 + kb + 16 + half * 8));
        v16h a1v = cat8(*(const v8h*)(arow1 + kb + half * 8),
                        *(const v8h*)(arow1 + kb + 16 + half * 8));

        const _Float16* bp = W2T + n * H1_ + kb + half * 8;   // L2-hot 64 KB
        v16h bm = cat8(*(const v8h*)bp, *(const v8h*)(bp + 16));

        acc0 = __builtin_amdgcn_wmma_f32_16x16x32_f16(false, a0,  false, bm,
                                                      (short)0, acc0, false, false);
        acc1 = __builtin_amdgcn_wmma_f32_16x16x32_f16(false, a1v, false, bm,
                                                      (short)0, acc1, false, false);
      }
      #pragma unroll
      for (int r = 0; r < 8; ++r) {
        int row0 = r + half * 8;
        int row1 = 16 + row0;
        float x;
        x = acc0[r]; x = (x >= 0.f) ? x : pa2 * x; sH2[row0 * H2_ + n] = (_Float16)x;
        x = acc1[r]; x = (x >= 0.f) ? x : pa2 * x; sH2[row1 * H2_ + n] = (_Float16)x;
      }
    }
    __syncthreads();

    // ---- layer 3: scores[32] = h2 @ W3 + b3, LDS f32 atomic reduction ----
    if (tid < 32) sAcc[tid] = pb3;
    __syncthreads();
    {
      int row = tid >> 3;              // 32 rows x 8 threads
      int c16 = (tid & 7) * 16;
      float p = 0.f;
      #pragma unroll
      for (int j = 0; j < 16; ++j)
        p += (float)sH2[row * H2_ + c16 + j] * W3[c16 + j];
      atomicAdd(&sAcc[row], p);
    }
    __syncthreads();
    if (tid < 32) {
      int l = l0 + tid;
      float s = sAcc[tid];
      bool keep = (l < L_) && (mask[(size_t)b * L_ + l] > 0);
      sScr[l] = keep ? s : NEGV;
    }
    __syncthreads();
  } // double row tiles

  // ---- masked softmax over L (wave32 shfl reductions) ----
  float m = (tid < LP_) ? sScr[tid] : NEGV;
  #pragma unroll
  for (int off = 16; off > 0; off >>= 1)
    m = fmaxf(m, __shfl_xor(m, off, 32));
  if (lane == 0) sRed[wave] = m;
  __syncthreads();
  if (tid == 0) {
    float mm = sRed[0];
    for (int w = 1; w < 8; ++w) mm = fmaxf(mm, sRed[w]);
    sRed[8] = mm;
  }
  __syncthreads();
  const float mglob = sRed[8];

  float e = 0.f;
  if (tid < L_) {
    float s = sScr[tid];
    if (s > -9999.0f) e = expf(s - mglob);  // masked entries skip expf and stay exactly 0
  }
  if (tid < LP_) sPrb[tid] = e;
  float ssum = e;
  #pragma unroll
  for (int off = 16; off > 0; off >>= 1)
    ssum += __shfl_xor(ssum, off, 32);
  if (lane == 0) sRed[wave] = ssum;
  __syncthreads();
  if (tid == 0) {
    float tt = 0.f;
    for (int w = 0; w < 8; ++w) tt += sRed[w];
    sRed[8] = tt;
  }
  __syncthreads();
  const float denom = sRed[8];
  const float inv = (denom > 0.f) ? (1.0f / denom) : 0.f;
  if (tid < LP_) sPrb[tid] = sPrb[tid] * inv;
  __syncthreads();

  // ---- user_interest[b][d] = sum_l probs[l] * keys[b][l][d] ----
  {
    int d  = tid & 127;
    int hh = tid >> 7;                        // split L in two halves for ILP
    float acc = 0.f;
    const float* kp = keys + ((size_t)b * L_) * D_ + d;
    for (int l = hh * 100; l < hh * 100 + 100; ++l)
      acc += sPrb[l] * kp[(size_t)l * D_];
    float* part = (float*)sAtt;               // reuse LDS
    part[hh * 128 + d] = acc;
  }
  __syncthreads();
  if (tid < D_) {
    float* part = (float*)sAtt;
    out[(size_t)b * D_ + tid] = part[tid] + part[128 + tid];
  }
}

// ---------------------------------------------------------------------------
extern "C" void kernel_launch(void* const* d_in, const int* in_sizes, int n_in,
                              void* d_out, int out_size, void* d_ws, size_t ws_size,
                              hipStream_t stream) {
  (void)in_sizes; (void)n_in; (void)out_size; (void)ws_size;
  const float* query = (const float*)d_in[0];
  const float* keys  = (const float*)d_in[1];
  const int*   maskp = (const int*)d_in[2];
  const float* W1    = (const float*)d_in[3];
  const float* b1    = (const float*)d_in[4];
  const float* a1    = (const float*)d_in[5];
  const float* W2    = (const float*)d_in[6];
  const float* b2    = (const float*)d_in[7];
  const float* a2    = (const float*)d_in[8];
  const float* W3    = (const float*)d_in[9];
  const float* b3    = (const float*)d_in[10];
  float* out = (float*)d_out;

  _Float16* W1T = (_Float16*)d_ws;                                  // 262144 B
  _Float16* W2T = (_Float16*)((char*)d_ws + (size_t)K1_ * H1_ * 2); // +65536 B

  din_prep_weights<<<512, 256, 0, stream>>>(W1, W2, W1T, W2T);
  din_attention_kernel<<<B_, 256, 0, stream>>>(query, keys, maskp,
                                               b1, a1, b2, a2, W3, b3,
                                               W1T, W2T, out);
}
